// LCASparsifier_71622874628692
// MI455X (gfx1250) — compile-verified, matrix operand order checked by reading, not run on hardware
//
#include <hip/hip_runtime.h>
#include <hip/hip_bf16.h>

typedef __attribute__((ext_vector_type(16))) _Float16 v16h;
typedef __attribute__((ext_vector_type(8)))  _Float16 v8h;
typedef __attribute__((ext_vector_type(8)))  float    v8f;
typedef int v4i __attribute__((vector_size(4 * sizeof(int))));

#define DIM          512
#define MROWS        64          // rows of u/v/a per workgroup (LDS-resident)
#define THREADS      512         // 16 waves -> 4 waves per SIMD32
#define ETA_C        0.1f
#define KTILES       16          // 512 / 32 (K per WMMA)
#define NT_PER_WAVE  8           // N-tiles per wave (8 x 16 cols = 128 cols)
#define MATMUL_ITERS 9           // step 1 is the zero-state special case

// LDS layout strides (padded for bank-conflict-free access, 64 banks x 4B)
#define V_STRIDE     516         // f32; 2064B row = 4-bank stride
#define A_STRIDE     520         // f16; 1040B row = 4-bank stride
#define G_STRIDE     40          // f16;   80B row = 20-bank stride (16B-aligned)
#define SLAB_ELEMS   (DIM * G_STRIDE)   // one K-slab: 512 rows x 32 f16 (+pad)
#define NSLABS       3                  // triple buffer (stage distance 2)

#define V_BYTES   ((size_t)MROWS * V_STRIDE * sizeof(float))
#define A_BYTES   ((size_t)MROWS * A_STRIDE * sizeof(_Float16))
#define G_BYTES   ((size_t)NSLABS * SLAB_ELEMS * sizeof(_Float16))
#define SMEM_BYTES (V_BYTES + A_BYTES + G_BYTES)          // 321,536 B <= 320KB

// --- CDNA5 async global->LDS copy (no VGPR transit, tracked by ASYNCcnt) ----
#if defined(__has_builtin)
# if __has_builtin(__builtin_amdgcn_global_load_async_to_lds_b128)
#  define HAVE_ASYNC_COPY 1
# endif
#endif
#ifndef HAVE_ASYNC_COPY
# define HAVE_ASYNC_COPY 0
#endif

__device__ __forceinline__ void wait_async_le(int n) {   // n is compile-time 0/4
#if HAVE_ASYNC_COPY
# if defined(__has_builtin) && __has_builtin(__builtin_amdgcn_s_wait_asynccnt)
  if (n == 0) __builtin_amdgcn_s_wait_asynccnt(0);
  else        __builtin_amdgcn_s_wait_asynccnt(4);
# else
  if (n == 0) asm volatile("s_wait_asynccnt 0x0" ::: "memory");
  else        asm volatile("s_wait_asynccnt 0x4" ::: "memory");
# endif
#else
  (void)n;
#endif
}

// ---------------------------------------------------------------------------
// Prep: Gh = f16( 0.5*(G + G^T) with zero diagonal ).  Symmetric, so WMMA
// B tiles (K x N) are fetched as coalesced *rows* of Gh.
// ---------------------------------------------------------------------------
__global__ void lca_prep_g(const float* __restrict__ G_raw,
                           _Float16* __restrict__ Gh) {
  int idx = blockIdx.x * blockDim.x + threadIdx.x;   // 0 .. 512*512-1
  int i = idx >> 9, j = idx & (DIM - 1);
  float g = (i == j) ? 0.0f : 0.5f * (G_raw[i * DIM + j] + G_raw[j * DIM + i]);
  Gh[idx] = (_Float16)g;
}

// Cooperatively copy K-slab kt (Gh[:, kt*32 .. +32)) into an LDS slab buffer.
// 512 rows x 64B = 32KB; 2048 x 16B chunks over 512 threads (coalesced).
// Async path: data flows L2 -> LDS directly; 4 async instrs per wave.
__device__ __forceinline__ void stage_slab(const _Float16* __restrict__ Gh,
                                           _Float16* __restrict__ slab,
                                           int kt, int tid) {
  const int kb = kt * 32;
  #pragma unroll
  for (int s = 0; s < 4; ++s) {
    const int chunk = tid + THREADS * s;   // 0..2047
    const int n     = chunk >> 2;          // G row
    const int part  = (chunk & 3) * 8;     // f16 offset within the 32-wide slab
    const _Float16* g = Gh + (size_t)n * DIM + kb + part;
    _Float16*       l = slab + n * G_STRIDE + part;
#if HAVE_ASYNC_COPY
    __builtin_amdgcn_global_load_async_to_lds_b128((v4i*)g, (v4i*)l, 0, 0);
#else
    *(v8h*)l = *(const v8h*)g;
#endif
  }
}

// ---------------------------------------------------------------------------
// Persistent fused LCA: each workgroup owns 64 rows. v (f32) and a (f16)
// live in LDS; G streams L2 -> triple-buffered LDS slab -> WMMA B operands.
// 16 waves: wave = (mt, nq) owns a 16-row M-tile x 8 N-tiles (acc = 64 VGPR)
// ---------------------------------------------------------------------------
__global__ __launch_bounds__(THREADS, 1) void lca_main(
    const float* __restrict__ u, const _Float16* __restrict__ Gh,
    const float* __restrict__ log_lam, float* __restrict__ out) {
  extern __shared__ char smem[];
  float*    vSh  = (float*)smem;
  _Float16* aSh  = (_Float16*)(smem + V_BYTES);
  _Float16* gSlb = (_Float16*)(smem + V_BYTES + A_BYTES); // [NSLABS][SLAB_ELEMS]

  const int tid      = threadIdx.x;
  const int lane     = tid & 31;          // wave32
  const int wv       = tid >> 5;          // 16 waves
  const int rowBlock = blockIdx.x * MROWS;
  const float lam    = __expf(log_lam[0]);

  // ---- step 1 (v = a = 0):  v = ETA*u ; a = soft(v, lam)
  for (int i = tid; i < MROWS * DIM; i += THREADS) {
    const int r = i >> 9, c = i & (DIM - 1);
    float uu = u[(size_t)rowBlock * DIM + i];
    float vv = ETA_C * uu;
    float m  = fabsf(vv) - lam;
    m = m > 0.0f ? m : 0.0f;
    vSh[r * V_STRIDE + c] = vv;
    aSh[r * A_STRIDE + c] = (_Float16)copysignf(m, vv);
  }
  stage_slab(Gh, gSlb,              0, tid);   // prime slabs for kt=0,1
  stage_slab(Gh, gSlb + SLAB_ELEMS, 1, tid);
  wait_async_le(0);
  __syncthreads();

  const int mt    = wv & 3;                  // M-tile (16 rows) of this wave
  const int nbase = (wv >> 2) * NT_PER_WAVE; // N-tile base: 0, 8, 16, 24
  const int half  = lane >> 4;               // 0: lanes 0-15, 1: lanes 16-31
  const int lrow  = lane & 15;

  const _Float16* aRow = aSh + (mt * 16 + lrow) * A_STRIDE;

  for (int it = 0; it < MATMUL_ITERS; ++it) {
    v8f acc[NT_PER_WAVE] = {};

    int rd = 0;                              // slab buffer read at this kt
    for (int kt = 0; kt < KTILES; ++kt) {
      // Stage slab kt+2 two phases ahead (buffer freed at end of kt-1).
      const bool issue = (kt + 2 < KTILES);
      if (issue) {
        int wr = rd + 2; if (wr >= NSLABS) wr -= NSLABS;
        stage_slab(Gh, gSlb + wr * SLAB_ELEMS, kt + 2, tid);
      }

      // 16-bit A layout: lanes 0-15 hold K[0..7] & K[16..23] of row M=lane,
      // lanes 16-31 hold K[8..15] & K[24..31] of row M=lane-16.
      const int kb = kt * 32 + half * 8;
      v8h a0 = *(const v8h*)(aRow + kb);        // ds_load_b128, conflict-free
      v8h a1 = *(const v8h*)(aRow + kb + 16);
      v16h afrag = __builtin_shufflevector(a0, a1,
          0,1,2,3,4,5,6,7,8,9,10,11,12,13,14,15);

      const _Float16* bs = gSlb + rd * SLAB_ELEMS + half * 8;
      #pragma unroll
      for (int j = 0; j < NT_PER_WAVE; ++j) {
        // B tile column n == row n of symmetric Gh, staged in the LDS slab.
        const _Float16* bRow = bs + ((nbase + j) * 16 + lrow) * G_STRIDE;
        v8h b0 = *(const v8h*)(bRow);
        v8h b1 = *(const v8h*)(bRow + 16);
        v16h bfrag = __builtin_shufflevector(b0, b1,
            0,1,2,3,4,5,6,7,8,9,10,11,12,13,14,15);
        acc[j] = __builtin_amdgcn_wmma_f32_16x16x32_f16(
            false, afrag, false, bfrag, (short)0, acc[j], false, false);
      }

      // In-order ASYNC completion: leaving <=4 outstanding means everything
      // issued before this kt has landed in LDS.
      if (issue) wait_async_le(4); else wait_async_le(0);
      __syncthreads();   // slab rotation + (at kt=15) all A reads complete
      rd = (rd + 1 == NSLABS) ? 0 : rd + 1;
    }

    // Fused update straight from the accumulators; a updated IN PLACE (all
    // matmul reads of aSh finished at the last kt barrier).
    // C layout: VGPR g -> M = g (lanes 0-15) / g+8 (lanes 16-31); N = lane&15.
    const bool last = (it == MATMUL_ITERS - 1);
    #pragma unroll
    for (int j = 0; j < NT_PER_WAVE; ++j) {
      const int c = (nbase + j) * 16 + lrow;
      #pragma unroll
      for (int g = 0; g < 8; ++g) {
        const int rl = mt * 16 + g + half * 8;
        float vv = vSh[rl * V_STRIDE + c];
        float uu = u[(size_t)(rowBlock + rl) * DIM + c];
        float vn = vv + ETA_C * (uu - vv - acc[j][g]);
        float m  = fabsf(vn) - lam;
        m = m > 0.0f ? m : 0.0f;
        float an = copysignf(m, vn);
        vSh[rl * V_STRIDE + c] = vn;
        aSh[rl * A_STRIDE + c] = (_Float16)an;
        if (last) out[(size_t)(rowBlock + rl) * DIM + c] = an;
      }
    }
    if (it + 1 < MATMUL_ITERS) {             // prime slabs 0,1 for next iter
      stage_slab(Gh, gSlb,              0, tid);
      stage_slab(Gh, gSlb + SLAB_ELEMS, 1, tid);
      wait_async_le(0);
    }
    __syncthreads();
  }
}

extern "C" void kernel_launch(void* const* d_in, const int* in_sizes, int n_in,
                              void* d_out, int out_size, void* d_ws, size_t ws_size,
                              hipStream_t stream) {
  const float* u       = (const float*)d_in[0];   // [65536, 512] f32
  const float* G_raw   = (const float*)d_in[1];   // [512, 512]  f32
  const float* log_lam = (const float*)d_in[2];   // scalar f32
  float*       out     = (float*)d_out;           // [65536, 512] f32
  _Float16*    Gh      = (_Float16*)d_ws;         // 512KB f16 scratch

  lca_prep_g<<<(DIM * DIM) / 256, 256, 0, stream>>>(G_raw, Gh);

  (void)hipFuncSetAttribute((const void*)lca_main,
                            hipFuncAttributeMaxDynamicSharedMemorySize,
                            (int)SMEM_BYTES);
  const int nRows   = in_sizes[0] / DIM;          // 65536
  const int nBlocks = nRows / MROWS;              // 1024
  lca_main<<<nBlocks, THREADS, SMEM_BYTES, stream>>>(u, Gh, log_lam, out);
}
